// BiMPM_77446850281509
// MI455X (gfx1250) — matrix operand order repaired and checked by reading, not compiled, not run on hardware
//
#include <hip/hip_runtime.h>
#include <hip/hip_bf16.h>
#include <math.h>

// ---------------------------------------------------------------------------
// BiMPM forward for MI455X (gfx1250, wave32, WMMA).
// All dense contractions use v_wmma_f32_16x16x32_f16 (f16 in, f32 accum).
// LSTM scans run as persistent 32-wave workgroups with h/c/gates in LDS;
// gate nonlinearities use hardware v_exp/v_rcp (TRANS pipe). Prefetches
// (global_prefetch_b8, WGP scope) are issued once per tile BEFORE the K-loops
// so the inner loops stay branch-free and fully pipelined.
// ---------------------------------------------------------------------------

typedef _Float16 half_t;
typedef __attribute__((ext_vector_type(16))) _Float16 v16h;
typedef __attribute__((ext_vector_type(8)))  float    v8f;

#define EPSV 1e-8f

// Problem constants
#define BB 32
#define TT 64
#define DD 300
#define DP 320      // D padded to multiple of 32
#define HH 256
#define GG 1024     // 4*H
#define LL 20
#define MVW 160     // 8*L
#define MVP 192     // MVW padded to multiple of 32
#define NROW 2048   // B*T

union FragH { v16h v; unsigned u[8]; half_t h[16]; };

// A-matrix 16x32 f16, row-major source: elem(m,k) at base[m*ld + k].
// Layout used here: lanes 0-15 -> M, kb=0; lanes 16-31 -> M, kb=16.
// VGPR p: k = kb + 2*p (contiguous K per lane half).
__device__ inline v16h load_a_rowmajor(const half_t* base, int ld) {
  int lane = threadIdx.x & 31;
  int m  = lane & 15;
  int kb = (lane & 16) ? 16 : 0;
  FragH f;
  const half_t* row = base + m * ld;
#pragma unroll
  for (int g = 0; g < 2; ++g)
#pragma unroll
    for (int j = 0; j < 4; ++j) {
      int k = kb + g * 8 + 2 * j;
      f.u[g * 4 + j] = *reinterpret_cast<const unsigned*>(row + k);
    }
  return f.v;
}

// B-matrix 32x16 f16 where memory holds W (N rows x K cols) and B = W^T:
// B[k,n] = W[n,k]. Lane: n = lane&15, kb = (lane&16)?16:0; VGPR p: k = kb+2p.
__device__ inline v16h load_b_from_wT(const half_t* Wrow0, int ld) {
  int lane = threadIdx.x & 31;
  int n  = lane & 15;
  int kb = (lane & 16) ? 16 : 0;
  FragH f;
  const half_t* row = Wrow0 + n * ld + kb;
#pragma unroll
  for (int p = 0; p < 8; ++p)
    f.u[p] = *reinterpret_cast<const unsigned*>(row + 2 * p);
  return f.v;
}

// B-matrix 32x16 f16 where memory is K-major (K rows x N cols): elem(k,n) at base[k*ld+n].
__device__ inline v16h load_b_kmajor(const half_t* base, int ld) {
  int lane = threadIdx.x & 31;
  int n  = lane & 15;
  int kb = (lane & 16) ? 16 : 0;
  FragH f;
#pragma unroll
  for (int p = 0; p < 8; ++p) {
    f.h[2 * p]     = base[(kb + 2 * p) * ld + n];
    f.h[2 * p + 1] = base[(kb + 2 * p + 1) * ld + n];
  }
  return f.v;
}

__device__ inline v8f wmma_f16(v16h a, v16h b, v8f c) {
  return __builtin_amdgcn_wmma_f32_16x16x32_f16(false, a, false, b, (short)0, c, false, false);
}

#define LOG2E 1.4426950408889634f

// Fast sigmoid/tanh on the TRANS pipe (v_exp_f32 + v_rcp_f32).
__device__ inline float sigm(float x) {
  return __builtin_amdgcn_rcpf(1.f + __builtin_amdgcn_exp2f(-LOG2E * x));
}
__device__ inline float tanh_fast(float x) {
  return 1.f - 2.f * __builtin_amdgcn_rcpf(__builtin_amdgcn_exp2f(2.f * LOG2E * x) + 1.f);
}
__device__ inline float divs(float n, float d) { return n / (d > EPSV ? d : EPSV); }

// Prefetch a row of `bytes` bytes at WGP scope (locality 3 -> nearest cache).
__device__ inline void prefetch_row(const void* p, int bytes) {
  const char* c = (const char*)p;
  for (int o = 0; o < bytes; o += 128) __builtin_prefetch(c + o, 0, 3);
}

// ---------------------------------------------------------------------------
// Prep kernels
// ---------------------------------------------------------------------------
__global__ void k_gather(const int* tok, const float* emb, half_t* x, int nrows) {
  int total = nrows * DP;
  for (int e = blockIdx.x * blockDim.x + threadIdx.x; e < total; e += gridDim.x * blockDim.x) {
    int r = e / DP, c = e % DP;
    x[e] = (c < DD) ? (half_t)emb[(long)tok[r] * DD + c] : (half_t)0.f;
  }
}

__global__ void k_cast_pad(const float* src, half_t* dst, int rows, int cols, int ldd) {
  int total = rows * ldd;
  for (int e = blockIdx.x * blockDim.x + threadIdx.x; e < total; e += gridDim.x * blockDim.x) {
    int r = e / ldd, c = e % ldd;
    dst[e] = (c < cols) ? (half_t)src[r * cols + c] : (half_t)0.f;
  }
}

// ---------------------------------------------------------------------------
// Y(MxN) = X(MxK) @ W(NxK)^T + b1 + b2, f16 inputs, f32 out. One 16x16 tile/wave.
// Tile operands are prefetched in full before the branch-free K-loop.
// ---------------------------------------------------------------------------
__global__ __launch_bounds__(256) void k_gemm_bias(const half_t* X, const half_t* W,
                                                   const float* b1, const float* b2,
                                                   float* Y, int M, int N, int K) {
  int wave = (int)((blockIdx.x * blockDim.x + threadIdx.x) >> 5);
  int ntiles = N >> 4;
  int total = (M >> 4) * ntiles;
  if (wave >= total) return;                 // wave-uniform
  int mt = wave / ntiles, nt = wave % ntiles;
  const half_t* Xb = X + (long)mt * 16 * K;
  const half_t* Wb = W + (long)nt * 16 * K;
  // Warm the whole 16xK tiles (lanes 0..15 each take one row) before looping.
  {
    int r = threadIdx.x & 15;
    prefetch_row(Xb + (long)r * K, K * 2);
    prefetch_row(Wb + (long)r * K, K * 2);
  }
  v8f acc = {};
  for (int k = 0; k < K; k += 32) {
    v16h a = load_a_rowmajor(Xb + k, K);
    v16h b = load_b_from_wT(Wb + k, K);
    acc = wmma_f16(a, b, acc);
  }
  int lane = threadIdx.x & 31;
  int n = nt * 16 + (lane & 15);
  int mb = (lane & 16) ? 8 : 0;
  float bias = b1[n] + b2[n];
#pragma unroll
  for (int r = 0; r < 8; ++r)
    Y[(long)(mt * 16 + mb + r) * N + n] = acc[r] + bias;
}

// ---------------------------------------------------------------------------
// Persistent LSTM scan: one block per (sequence,direction). 32 waves.
// Per step: gates(32x1024) = h(32x256) @ whh^T via 128 WMMA tiles (4/wave),
// then elementwise cell update. h(f16), c(f32), gates(f32) live in LDS.
// whh (512KB f16) is warmed once before the time loop; steps re-hit cache.
// ---------------------------------------------------------------------------
struct ScanArgs {
  const float*  xg;    // (B*T, 1024) precomputed x@wih^T + biases
  const half_t* whh;   // (1024, 256) f16
  float*        hs;    // (B,T,H) f32 outputs (time-corrected order)
  half_t*       hs16;  // f16 copy for downstream WMMA
  float*        hT;    // (B,H) final hidden
  int           rev;
};
struct ScanBatch { ScanArgs a[4]; };

__global__ __launch_bounds__(1024) void k_lstm_scan(ScanBatch sb) {
  ScanArgs A = sb.a[blockIdx.x];
  extern __shared__ char smem[];
  half_t* h_lds = (half_t*)smem;                              // 32*256 f16 = 16KB
  float*  c_st  = (float*)(smem + BB * HH * 2);               // 32*256 f32 = 32KB
  float*  g_lds = (float*)(smem + BB * HH * 2 + BB * HH * 4); // 32*1024 f32 = 128KB
  int tid = threadIdx.x;
  // One-time warm of the recurrent weights (4 cachelines per thread).
  {
    const char* wp = (const char*)A.whh;
    for (int i = tid; i < (GG * HH * 2) / 128; i += 1024)
      __builtin_prefetch(wp + (long)i * 128, 0, 3);
  }
  for (int e = tid; e < BB * HH; e += 1024) { h_lds[e] = (half_t)0.f; c_st[e] = 0.f; }
  __syncthreads();
  int wave = tid >> 5, lane = tid & 31;
  for (int s = 0; s < TT; ++s) {
    // phase 1: gate pre-activations from recurrent GEMM
#pragma unroll
    for (int q = 0; q < 4; ++q) {
      int tile = wave + 32 * q;          // 128 tiles: 2 mtiles x 64 ntiles
      int mt = tile >> 6, nt = tile & 63;
      const half_t* Wb = A.whh + (long)nt * 16 * HH;
      v8f acc = {};
#pragma unroll
      for (int kt = 0; kt < 8; ++kt) {
        v16h a = load_a_rowmajor(h_lds + mt * 16 * HH + kt * 32, HH);
        v16h b = load_b_from_wT(Wb + kt * 32, HH);
        acc = wmma_f16(a, b, acc);
      }
      int n = nt * 16 + (lane & 15);
      int mb = (lane & 16) ? 8 : 0;
#pragma unroll
      for (int r = 0; r < 8; ++r) g_lds[(mt * 16 + mb + r) * GG + n] = acc[r];
    }
    __syncthreads();
    // phase 2: cell update (TRANS-pipe sigmoid/tanh)
    int tt = A.rev ? (TT - 1 - s) : s;
    for (int e = tid; e < BB * HH; e += 1024) {
      int bi = e >> 8, hh = e & 255;
      const float* xrow = A.xg + (long)(bi * TT + tt) * GG;
      float gi = g_lds[bi * GG + hh]        + xrow[hh];
      float gf = g_lds[bi * GG + 256 + hh]  + xrow[256 + hh];
      float gg = g_lds[bi * GG + 512 + hh]  + xrow[512 + hh];
      float go = g_lds[bi * GG + 768 + hh]  + xrow[768 + hh];
      float c = sigm(gf) * c_st[e] + sigm(gi) * tanh_fast(gg);
      float h = sigm(go) * tanh_fast(c);
      c_st[e] = c;
      h_lds[e] = (half_t)h;
      long off = (long)(bi * TT + tt) * HH + hh;
      A.hs[off] = h;
      A.hs16[off] = (half_t)h;
      if (s == TT - 1) A.hT[bi * HH + hh] = h;
    }
    __syncthreads();
  }
}

// ---------------------------------------------------------------------------
// Attention: att[b,i,j] = divs(vp_i . vh_j, |vp_i||vh_j|). 16 waves = 4x4 tiles.
// Also emits att^T, f16 copies, row/col sums.
// ---------------------------------------------------------------------------
__global__ __launch_bounds__(512) void k_attention(const half_t* v1h, const half_t* v2h,
                                                   const float* v1, const float* v2,
                                                   float* att, float* attT,
                                                   half_t* att16, half_t* att16t,
                                                   float* rowsum, float* colsum) {
  __shared__ float c_lds[TT * TT];
  __shared__ float np[TT], nh[TT];
  int b = blockIdx.x, tid = threadIdx.x;
  if (tid < 128) {
    int r = tid & 63;
    const float* src = (tid < 64 ? v1 : v2) + (long)(b * TT + r) * HH;
    float s = 0.f;
    for (int h = 0; h < HH; ++h) { float x = src[h]; s += x * x; }
    (tid < 64 ? np : nh)[r] = sqrtf(s);
  }
  int wave = tid >> 5, lane = tid & 31;
  int mt = wave >> 2, nt = wave & 3;
  v8f acc = {};
#pragma unroll
  for (int kt = 0; kt < 8; ++kt) {
    v16h a = load_a_rowmajor(v1h + (long)b * TT * HH + mt * 16 * HH + kt * 32, HH);
    v16h bf = load_b_from_wT(v2h + (long)b * TT * HH + nt * 16 * HH + kt * 32, HH);
    acc = wmma_f16(a, bf, acc);
  }
  { int n = nt * 16 + (lane & 15); int mb = (lane & 16) ? 8 : 0;
#pragma unroll
    for (int r = 0; r < 8; ++r) c_lds[(mt * 16 + mb + r) * TT + n] = acc[r]; }
  __syncthreads();
  for (int e = tid; e < TT * TT; e += 512) {
    int i = e >> 6, j = e & 63;
    float a = divs(c_lds[e], np[i] * nh[j]);
    c_lds[e] = a;
    att[(long)b * TT * TT + e]                 = a;
    attT[(long)b * TT * TT + j * TT + i]       = a;
    att16[(long)b * TT * TT + e]               = (half_t)a;
    att16t[(long)b * TT * TT + j * TT + i]     = (half_t)a;
  }
  __syncthreads();
  if (tid < 128) {
    int r = tid & 63; float s = 0.f;
    if (tid < 64) { for (int j = 0; j < TT; ++j) s += c_lds[r * TT + j]; rowsum[b * TT + r] = s; }
    else          { for (int i = 0; i < TT; ++i) s += c_lds[i * TT + r]; colsum[b * TT + r] = s; }
  }
}

// out[b,m,:] = (A16[b] (64x64) @ V16[b] (64x256)) / divs-denom[b,m]. 64 tiles, 4/wave.
__global__ __launch_bounds__(512) void k_am(const half_t* A16, const half_t* V16,
                                            const float* denom, float* out) {
  int b = blockIdx.x, tid = threadIdx.x;
  int wave = tid >> 5, lane = tid & 31;
#pragma unroll
  for (int q = 0; q < 4; ++q) {
    int tile = wave + 16 * q;      // 4 mtiles x 16 ntiles
    int mt = tile >> 4, nt = tile & 15;
    v8f acc = {};
#pragma unroll
    for (int kt = 0; kt < 2; ++kt) {
      v16h a = load_a_rowmajor(A16 + (long)b * TT * TT + mt * 16 * TT + kt * 32, TT);
      v16h bf = load_b_kmajor(V16 + (long)b * TT * HH + kt * 32 * HH + nt * 16, HH);
      acc = wmma_f16(a, bf, acc);
    }
    int n = nt * 16 + (lane & 15);
    int mb = (lane & 16) ? 8 : 0;
#pragma unroll
    for (int r = 0; r < 8; ++r) {
      int m = mt * 16 + mb + r;
      out[(long)(b * TT + m) * HH + n] = divs(acc[r], denom[b * TT + m]);
    }
  }
}

// out[b,r,h] = max_s A[b,r,s] * V[b,s,h]
__global__ __launch_bounds__(256) void k_ax(const float* A, const float* V, float* out) {
  __shared__ float ar[TT];
  int bx = blockIdx.x; int b = bx >> 6, r = bx & 63;
  int tid = threadIdx.x;
  if (tid < TT) ar[tid] = A[(long)b * TT * TT + r * TT + tid];
  __syncthreads();
  float m = -1e30f;
  for (int s = 0; s < TT; ++s) m = fmaxf(m, ar[s] * V[(long)(b * TT + s) * HH + tid]);
  out[(long)(b * TT + r) * HH + tid] = m;
}

// mp_full: out[b,t,col+l] = num / max(n1*n2, EPS), l in [0,20)
__global__ __launch_bounds__(256) void k_mp_full(const float* v1, const float* v2base,
                                                 long v2_bs, long v2_ts,
                                                 const float* w, float* out, int col) {
  __shared__ float v1r[HH], v2r[HH];
  int bx = blockIdx.x; int b = bx >> 6, t = bx & 63;
  int tid = threadIdx.x;
  v1r[tid] = v1[(long)(b * TT + t) * HH + tid];
  v2r[tid] = v2base[(long)b * v2_bs + (long)t * v2_ts + tid];
  __syncthreads();
  int wave = tid >> 5, lane = tid & 31;
  for (int l = wave; l < LL; l += 8) {
    const float* wl = w + l * HH;
    float num = 0.f, a1 = 0.f, a2 = 0.f;
#pragma unroll
    for (int k = 0; k < 8; ++k) {
      int h = lane + 32 * k;
      float w2 = wl[h] * wl[h];
      num += v1r[h] * w2 * v2r[h];
      a1  += v1r[h] * v1r[h] * w2;
      a2  += v2r[h] * v2r[h] * w2;
    }
    for (int off = 16; off; off >>= 1) {
      num += __shfl_xor(num, off, 32);
      a1  += __shfl_xor(a1, off, 32);
      a2  += __shfl_xor(a2, off, 32);
    }
    if (lane == 0) {
      float d = sqrtf(a1) * sqrtf(a2);
      out[(long)(b * TT + t) * MVW + col + l] = num / fmaxf(d, EPSV);
    }
  }
}

// mp_pair (w_max): per (b,l), 64x64 GEMM of (v1*w_l) @ v2^T via WMMA; write row/col maxima.
__global__ __launch_bounds__(512) void k_pair(const float* v1, const float* v2,
                                              const half_t* v2h, const float* w,
                                              float* mvp, float* mvh, int col) {
  __shared__ half_t a_lds[TT * HH];   // 32KB
  __shared__ float  c_lds[TT * TT];   // 16KB
  __shared__ float  n1[TT], n2[TT];
  int b = blockIdx.x / LL, l = blockIdx.x % LL;
  int tid = threadIdx.x;
  const float* wl = w + l * HH;
  for (int e = tid; e < TT * HH; e += 512) {
    int i = e >> 8, h = e & 255;
    a_lds[e] = (half_t)(v1[(long)(b * TT + i) * HH + h] * wl[h]);
  }
  __syncthreads();
  if (tid < 128) {
    int r = tid & 63; float s = 0.f;
    if (tid < 64) { for (int h = 0; h < HH; ++h) { float x = (float)a_lds[r * HH + h]; s += x * x; } n1[r] = sqrtf(s); }
    else { for (int h = 0; h < HH; ++h) { float x = v2[(long)(b * TT + r) * HH + h] * wl[h]; s += x * x; } n2[r] = sqrtf(s); }
  }
  int wave = tid >> 5, lane = tid & 31;
  int mt = wave >> 2, nt = wave & 3;
  v8f acc = {};
#pragma unroll
  for (int kt = 0; kt < 8; ++kt) {
    v16h a = load_a_rowmajor(a_lds + mt * 16 * HH + kt * 32, HH);
    v16h bf = load_b_from_wT(v2h + (long)b * TT * HH + nt * 16 * HH + kt * 32, HH);
    acc = wmma_f16(a, bf, acc);
  }
  { int n = nt * 16 + (lane & 15); int mb = (lane & 16) ? 8 : 0;
#pragma unroll
    for (int r = 0; r < 8; ++r) c_lds[(mt * 16 + mb + r) * TT + n] = acc[r]; }
  __syncthreads();
  if (tid < 128) {
    int r = tid & 63;
    if (tid < 64) {
      float m = -1e30f;
      for (int j = 0; j < TT; ++j) m = fmaxf(m, divs(c_lds[r * TT + j], n1[r] * n2[j]));
      mvp[(long)(b * TT + r) * MVW + col] = m;
    } else {
      float m = -1e30f;
      for (int i = 0; i < TT; ++i) m = fmaxf(m, divs(c_lds[i * TT + r], n1[i] * n2[r]));
      mvh[(long)(b * TT + r) * MVW + col] = m;
    }
  }
}

// Final FC: out[b,:2] = fc2 @ tanh(fc1 @ concat(hpf,hpb,hhf,hhb) + b1) + b2
__global__ __launch_bounds__(512) void k_fc(const float* hpf, const float* hpb,
                                            const float* hhf, const float* hhb,
                                            const float* fc1w, const float* fc1b,
                                            const float* fc2w, const float* fc2b,
                                            float* out) {
  __shared__ float x[1024];
  __shared__ float y[512];
  int b = blockIdx.x, tid = threadIdx.x;
  for (int j = tid; j < 1024; j += 512) {
    int seg = j >> 8, k = j & 255;
    const float* src = (seg == 0) ? hpf : (seg == 1) ? hpb : (seg == 2) ? hhf : hhb;
    x[j] = src[b * HH + k];
  }
  __syncthreads();
  float s = fc1b[tid];
  for (int k = 0; k < 1024; ++k) s += fc1w[(long)tid * 1024 + k] * x[k];
  y[tid] = tanhf(s);
  __syncthreads();
  if (tid < 2) {
    float o = fc2b[tid];
    for (int k = 0; k < 512; ++k) o += fc2w[tid * 512 + k] * y[k];
    out[b * 2 + tid] = o;
  }
}

// ---------------------------------------------------------------------------
// Host-side orchestration
// ---------------------------------------------------------------------------
extern "C" void kernel_launch(void* const* d_in, const int* in_sizes, int n_in,
                              void* d_out, int out_size, void* d_ws, size_t ws_size,
                              hipStream_t stream) {
  (void)in_sizes; (void)n_in; (void)out_size; (void)ws_size;

  const int*   q1      = (const int*)d_in[0];
  const int*   q2      = (const int*)d_in[1];
  const float* emb     = (const float*)d_in[4];
  const float* cwih_f  = (const float*)d_in[5];
  const float* cwhh_f  = (const float*)d_in[6];
  const float* cbih_f  = (const float*)d_in[7];
  const float* cbhh_f  = (const float*)d_in[8];
  const float* cwih_b  = (const float*)d_in[9];
  const float* cwhh_b  = (const float*)d_in[10];
  const float* cbih_b  = (const float*)d_in[11];
  const float* cbhh_b  = (const float*)d_in[12];
  const float* mp_w    = (const float*)d_in[13];
  const float* awih_f  = (const float*)d_in[14];
  const float* awhh_f  = (const float*)d_in[15];
  const float* abih_f  = (const float*)d_in[16];
  const float* abhh_f  = (const float*)d_in[17];
  const float* awih_b  = (const float*)d_in[18];
  const float* awhh_b  = (const float*)d_in[19];
  const float* abih_b  = (const float*)d_in[20];
  const float* abhh_b  = (const float*)d_in[21];
  const float* fc1w    = (const float*)d_in[22];
  const float* fc1b    = (const float*)d_in[23];
  const float* fc2w    = (const float*)d_in[24];
  const float* fc2b    = (const float*)d_in[25];
  float* out = (float*)d_out;

  // Bump allocator over workspace (deterministic layout every call).
  char* ws = (char*)d_ws;
  size_t off = 0;
  auto alloc = [&](size_t bytes) -> void* {
    off = (off + 255) & ~(size_t)255;
    void* p = ws + off;
    off += bytes;
    return p;
  };

  half_t* xq1h   = (half_t*)alloc((size_t)NROW * DP * 2);
  half_t* xq2h   = (half_t*)alloc((size_t)NROW * DP * 2);
  half_t* wihFh  = (half_t*)alloc((size_t)GG * DP * 2);
  half_t* wihBh  = (half_t*)alloc((size_t)GG * DP * 2);
  half_t* whhFh  = (half_t*)alloc((size_t)GG * HH * 2);
  half_t* whhBh  = (half_t*)alloc((size_t)GG * HH * 2);
  half_t* awihFh = (half_t*)alloc((size_t)GG * MVP * 2);
  half_t* awihBh = (half_t*)alloc((size_t)GG * MVP * 2);
  half_t* awhhFh = (half_t*)alloc((size_t)GG * HH * 2);
  half_t* awhhBh = (half_t*)alloc((size_t)GG * HH * 2);
  half_t* hs16[4];
  for (int i = 0; i < 4; ++i) hs16[i] = (half_t*)alloc((size_t)NROW * HH * 2);
  half_t* mvp16  = (half_t*)alloc((size_t)NROW * MVP * 2);
  half_t* mvh16  = (half_t*)alloc((size_t)NROW * MVP * 2);
  half_t* att16  = (half_t*)alloc((size_t)BB * TT * TT * 2);
  half_t* att16t = (half_t*)alloc((size_t)BB * TT * TT * 2);

  float* xg[4];
  for (int i = 0; i < 4; ++i) xg[i] = (float*)alloc((size_t)NROW * GG * 4);
  float* hs32[4];  // p_fw, p_bw, h_fw, h_bw
  for (int i = 0; i < 4; ++i) hs32[i] = (float*)alloc((size_t)NROW * HH * 4);
  float* mvp    = (float*)alloc((size_t)NROW * MVW * 4);
  float* mvh    = (float*)alloc((size_t)NROW * MVW * 4);
  float* att    = (float*)alloc((size_t)BB * TT * TT * 4);
  float* attT   = (float*)alloc((size_t)BB * TT * TT * 4);
  float* rowsum = (float*)alloc((size_t)BB * TT * 4);
  float* colsum = (float*)alloc((size_t)BB * TT * 4);
  float* amh    = (float*)alloc((size_t)NROW * HH * 4);
  float* amp    = (float*)alloc((size_t)NROW * HH * 4);
  float* axh    = (float*)alloc((size_t)NROW * HH * 4);
  float* axp    = (float*)alloc((size_t)NROW * HH * 4);
  float* hT[4];
  for (int i = 0; i < 4; ++i) hT[i] = (float*)alloc((size_t)BB * HH * 4);

  // --- 1. Embedding gather + weight casts ---
  k_gather<<<(NROW * DP + 255) / 256, 256, 0, stream>>>(q1, emb, xq1h, NROW);
  k_gather<<<(NROW * DP + 255) / 256, 256, 0, stream>>>(q2, emb, xq2h, NROW);
  k_cast_pad<<<(GG * DP + 255) / 256, 256, 0, stream>>>(cwih_f, wihFh, GG, DD, DP);
  k_cast_pad<<<(GG * DP + 255) / 256, 256, 0, stream>>>(cwih_b, wihBh, GG, DD, DP);
  k_cast_pad<<<(GG * HH + 255) / 256, 256, 0, stream>>>(cwhh_f, whhFh, GG, HH, HH);
  k_cast_pad<<<(GG * HH + 255) / 256, 256, 0, stream>>>(cwhh_b, whhBh, GG, HH, HH);
  k_cast_pad<<<(GG * MVP + 255) / 256, 256, 0, stream>>>(awih_f, awihFh, GG, MVW, MVP);
  k_cast_pad<<<(GG * MVP + 255) / 256, 256, 0, stream>>>(awih_b, awihBh, GG, MVW, MVP);
  k_cast_pad<<<(GG * HH + 255) / 256, 256, 0, stream>>>(awhh_f, awhhFh, GG, HH, HH);
  k_cast_pad<<<(GG * HH + 255) / 256, 256, 0, stream>>>(awhh_b, awhhBh, GG, HH, HH);

  // --- 2. Context xg GEMMs (2048x1024, K=320), WMMA ---
  int gemm_blocks = (NROW / 16) * (GG / 16) / 8;   // 1024
  k_gemm_bias<<<gemm_blocks, 256, 0, stream>>>(xq1h, wihFh, cbih_f, cbhh_f, xg[0], NROW, GG, DP);
  k_gemm_bias<<<gemm_blocks, 256, 0, stream>>>(xq1h, wihBh, cbih_b, cbhh_b, xg[1], NROW, GG, DP);
  k_gemm_bias<<<gemm_blocks, 256, 0, stream>>>(xq2h, wihFh, cbih_f, cbhh_f, xg[2], NROW, GG, DP);
  k_gemm_bias<<<gemm_blocks, 256, 0, stream>>>(xq2h, wihBh, cbih_b, cbhh_b, xg[3], NROW, GG, DP);

  // --- 3. Context BiLSTM scan (persistent, 4 blocks x 32 waves) ---
  const size_t scan_lds = (size_t)BB * HH * 2 + (size_t)BB * HH * 4 + (size_t)BB * GG * 4; // 180224
  {
    ScanBatch sb;
    sb.a[0] = { xg[0], whhFh, hs32[0], hs16[0], hT[0], 0 };  // p_fw
    sb.a[1] = { xg[1], whhBh, hs32[1], hs16[1], hT[1], 1 };  // p_bw
    sb.a[2] = { xg[2], whhFh, hs32[2], hs16[2], hT[2], 0 };  // h_fw
    sb.a[3] = { xg[3], whhBh, hs32[3], hs16[3], hT[3], 1 };  // h_bw
    k_lstm_scan<<<4, 1024, scan_lds, stream>>>(sb);
  }

  // --- 4. Matching, per direction ---
  for (int d = 0; d < 2; ++d) {
    const float*  vp32 = hs32[d];          // p_fw / p_bw
    const float*  vh32 = hs32[2 + d];      // h_fw / h_bw
    const half_t* vp16 = hs16[d];
    const half_t* vh16 = hs16[2 + d];
    int idx = d ? 0 : (TT - 1);
    const float* wfull = mp_w + (0 + d) * LL * HH;
    const float* wmax  = mp_w + (2 + d) * LL * HH;
    const float* wam   = mp_w + (4 + d) * LL * HH;
    const float* wax   = mp_w + (6 + d) * LL * HH;
    int colb = d * 80;

    k_attention<<<BB, 512, 0, stream>>>(vp16, vh16, vp32, vh32, att, attT,
                                        att16, att16t, rowsum, colsum);
    k_am<<<BB, 512, 0, stream>>>(att16, vh16, rowsum, amh);   // mean of vh weighted by att rows
    k_am<<<BB, 512, 0, stream>>>(att16t, vp16, colsum, amp);  // mean of vp weighted by att cols
    k_ax<<<BB * TT, 256, 0, stream>>>(att,  vh32, axh);
    k_ax<<<BB * TT, 256, 0, stream>>>(attT, vp32, axp);

    // full-match (v2 = other side at idx; broadcast over t)
    k_mp_full<<<BB * TT, 256, 0, stream>>>(vp32, vh32 + (long)idx * HH, (long)TT * HH, 0L,
                                           wfull, mvp, colb + 0);
    k_mp_full<<<BB * TT, 256, 0, stream>>>(vh32, vp32 + (long)idx * HH, (long)TT * HH, 0L,
                                           wfull, mvh, colb + 0);
    // max-match
    k_pair<<<BB * LL, 512, 0, stream>>>(vp32, vh32, vh16, wmax, mvp, mvh, colb + 20);
    // attentive-mean match
    k_mp_full<<<BB * TT, 256, 0, stream>>>(vp32, amh, (long)TT * HH, (long)HH, wam, mvp, colb + 40);
    k_mp_full<<<BB * TT, 256, 0, stream>>>(vh32, amp, (long)TT * HH, (long)HH, wam, mvh, colb + 40);
    // attentive-max match
    k_mp_full<<<BB * TT, 256, 0, stream>>>(vp32, axh, (long)TT * HH, (long)HH, wax, mvp, colb + 60);
    k_mp_full<<<BB * TT, 256, 0, stream>>>(vh32, axp, (long)TT * HH, (long)HH, wax, mvh, colb + 60);
  }

  // --- 5. Aggregation BiLSTM ---
  k_cast_pad<<<(NROW * MVP + 255) / 256, 256, 0, stream>>>(mvp, mvp16, NROW, MVW, MVP);
  k_cast_pad<<<(NROW * MVP + 255) / 256, 256, 0, stream>>>(mvh, mvh16, NROW, MVW, MVP);
  k_gemm_bias<<<gemm_blocks, 256, 0, stream>>>(mvp16, awihFh, abih_f, abhh_f, xg[0], NROW, GG, MVP);
  k_gemm_bias<<<gemm_blocks, 256, 0, stream>>>(mvp16, awihBh, abih_b, abhh_b, xg[1], NROW, GG, MVP);
  k_gemm_bias<<<gemm_blocks, 256, 0, stream>>>(mvh16, awihFh, abih_f, abhh_f, xg[2], NROW, GG, MVP);
  k_gemm_bias<<<gemm_blocks, 256, 0, stream>>>(mvh16, awihBh, abih_b, abhh_b, xg[3], NROW, GG, MVP);
  {
    ScanBatch sb;  // hs buffers reused as scratch; hT are the real outputs
    sb.a[0] = { xg[0], awhhFh, hs32[0], hs16[0], hT[0], 0 };  // hpf
    sb.a[1] = { xg[1], awhhBh, hs32[1], hs16[1], hT[1], 1 };  // hpb
    sb.a[2] = { xg[2], awhhFh, hs32[2], hs16[2], hT[2], 0 };  // hhf
    sb.a[3] = { xg[3], awhhBh, hs32[3], hs16[3], hT[3], 1 };  // hhb
    k_lstm_scan<<<4, 1024, scan_lds, stream>>>(sb);
  }

  // --- 6. Final FC ---
  k_fc<<<BB, 512, 0, stream>>>(hT[0], hT[1], hT[2], hT[3], fc1w, fc1b, fc2w, fc2b, out);
}
